// STICKYLlamaAttention_50431505989664
// MI455X (gfx1250) — matrix operand order repaired and checked by reading, not compiled, not run on hardware
//
#include <hip/hip_runtime.h>
#include <hip/hip_bf16.h>

// ---------------------------------------------------------------------------
// MI455X (gfx1250, wave32) Llama attention prefill, bf16 WMMA pipeline.
// D = A x B + C via v_wmma_f32_16x16x32_bf16 throughout; global_prefetch_b8
// used to stream GEMM operands through the 192 MB L2.
// ---------------------------------------------------------------------------

#define SEQ   2048
#define HID   4096
#define NH    32
#define NKV   8
#define HD    128
#define KVP   1024   // NKV*HD

typedef __bf16          v16bf __attribute__((ext_vector_type(16)));
typedef float           v8f   __attribute__((ext_vector_type(8)));
typedef unsigned short  v8u   __attribute__((ext_vector_type(8)));
typedef unsigned short  v16u  __attribute__((ext_vector_type(16)));

__device__ __forceinline__ unsigned short f2bf(float f) {
    unsigned int u = __builtin_bit_cast(unsigned int, f);
    u += 0x7fffu + ((u >> 16) & 1u);          // round-to-nearest-even
    return (unsigned short)(u >> 16);
}
__device__ __forceinline__ float bf2f(unsigned short h) {
    unsigned int u = ((unsigned int)h) << 16;
    return __builtin_bit_cast(float, u);
}
__device__ __forceinline__ v8f wmma_bf16(v16bf a, v16bf b, v8f c) {
    return __builtin_amdgcn_wmma_f32_16x16x32_bf16(
        /*neg_a=*/false, a, /*neg_b=*/false, b,
        /*c_mod=*/(short)0, c, /*reuse_a=*/false, /*reuse_b=*/false);
}
// Build an A-fragment (16x32 bf16) from two contiguous 8-element chunks.
__device__ __forceinline__ v16bf make_afrag(const unsigned short* p) {
    v8u lo = *(const v8u*)p;
    v8u hi = *(const v8u*)(p + 8);
    v16u t;
#pragma unroll
    for (int e = 0; e < 8; ++e) { t[e] = lo[e]; t[e + 8] = hi[e]; }
    return __builtin_bit_cast(v16bf, t);
}
__device__ __forceinline__ v16bf make_bfrag(const unsigned short* p) {
    return __builtin_bit_cast(v16bf, *(const v16u*)p);
}

__device__ __forceinline__ void stout(float* p, float v)          { *p = v; }
__device__ __forceinline__ void stout(unsigned short* p, float v) { *p = f2bf(v); }

// ---------------------------------------------------------------------------
// fp32 -> bf16 cast, 8 elements/thread (all our sizes are multiples of 8)
// ---------------------------------------------------------------------------
__global__ void cvt_bf16_kernel(const float* __restrict__ src,
                                unsigned short* __restrict__ dst, int n8) {
    int i = blockIdx.x * blockDim.x + threadIdx.x;
    if (i >= n8) return;
    float4 a = ((const float4*)src)[i * 2];
    float4 b = ((const float4*)src)[i * 2 + 1];
    v8u o;
    o[0] = f2bf(a.x); o[1] = f2bf(a.y); o[2] = f2bf(a.z); o[3] = f2bf(a.w);
    o[4] = f2bf(b.x); o[5] = f2bf(b.y); o[6] = f2bf(b.z); o[7] = f2bf(b.w);
    *(v8u*)(dst + (size_t)i * 8) = o;
}

// ---------------------------------------------------------------------------
// C[M,N] = A[M,K] (bf16, row-major) x W[N,K]^T (bf16, row-major)
// 4 waves/block, each wave owns a 64x64 tile (4x4 WMMA accumulators).
// All operands are K-contiguous -> fragment loads are plain b128s; the 192 MB
// L2 provides tile reuse; global_prefetch_b8 pre-touches the next k-step.
// ---------------------------------------------------------------------------
template <typename OUT>
__global__ __launch_bounds__(128)
void gemm_bf16_wmma(const unsigned short* __restrict__ A,
                    const unsigned short* __restrict__ W,
                    OUT* __restrict__ C, int M, int N, int K) {
    const int lane = threadIdx.x & 31;
    const int wave = threadIdx.x >> 5;
    const int hf   = lane >> 4;      // wave half (0/1)
    const int lm   = lane & 15;
    const int tm   = blockIdx.x * 128 + (wave >> 1) * 64;
    const int tn   = blockIdx.y * 128 + (wave & 1) * 64;

    v8f acc[4][4];
#pragma unroll
    for (int i = 0; i < 4; ++i)
#pragma unroll
        for (int j = 0; j < 4; ++j) acc[i][j] = v8f{0.f,0.f,0.f,0.f,0.f,0.f,0.f,0.f};

    for (int k0 = 0; k0 < K; k0 += 32) {
        v16bf aF[4], bF[4];
#pragma unroll
        for (int i = 0; i < 4; ++i) {
            // A-fragment: row = tm+i*16+lm, k contiguous at k0+hf*16
            aF[i] = make_afrag(A + (size_t)(tm + i * 16 + lm) * K + k0 + hf * 16);
        }
#pragma unroll
        for (int j = 0; j < 4; ++j) {
            // B-fragment: n = tn+j*16+lm (row of W), k contiguous at k0+hf*16
            bF[j] = make_bfrag(W + (size_t)(tn + j * 16 + lm) * K + k0 + hf * 16);
        }
        if (k0 + 32 < K) {   // pre-touch next k-step (-> global_prefetch_b8)
            __builtin_prefetch(A + (size_t)(tm + lm) * K + k0 + 32 + hf * 8, 0, 3);
            __builtin_prefetch(W + (size_t)(tn + lm) * K + k0 + 32 + hf * 16, 0, 3);
        }
#pragma unroll
        for (int i = 0; i < 4; ++i)
#pragma unroll
            for (int j = 0; j < 4; ++j)
                acc[i][j] = wmma_bf16(aF[i], bF[j], acc[i][j]);
    }

#pragma unroll
    for (int i = 0; i < 4; ++i)
#pragma unroll
        for (int j = 0; j < 4; ++j)
#pragma unroll
            for (int r = 0; r < 8; ++r) {
                int row = tm + i * 16 + hf * 8 + r;   // C layout: M = r + half*8
                int col = tn + j * 16 + lm;           // N = lane%16
                stout(C + (size_t)row * N + col, acc[i][j][r]);
            }
}

// ---------------------------------------------------------------------------
// In-place RoPE on bf16 [S, H*128]; fold softmax scale into Q.
// Each thread rotates the (d, d+64) pair for one (s, h, d<64).
// ---------------------------------------------------------------------------
__global__ void rope_kernel(unsigned short* __restrict__ X, int H, int pitch,
                            float scale, int total) {
    int idx = blockIdx.x * blockDim.x + threadIdx.x;
    if (idx >= total) return;
    int i = idx & 63;
    int h = (idx >> 6) % H;
    int s = idx / (64 * H);
    // inv_freq = 10000^(-i/64) = exp(-i * ln(10000)/64)
    float inv = __expf(-0.14391156f * (float)i);
    float ang = (float)s * inv;
    float c, sn;
    __sincosf(ang, &c, &sn);
    unsigned short* p = X + (size_t)s * pitch + h * HD + i;
    float x1 = bf2f(p[0]);
    float x2 = bf2f(p[64]);
    p[0]  = f2bf((x1 * c - x2 * sn) * scale);
    p[64] = f2bf((x2 * c + x1 * sn) * scale);
}

// ---------------------------------------------------------------------------
// V [s, kv*128+d] -> Vt [kv*128+d, s]  (so P.V B-fragments are key-contiguous)
// ---------------------------------------------------------------------------
__global__ void transpose_v_kernel(const unsigned short* __restrict__ V,
                                   unsigned short* __restrict__ Vt) {
    int idx = blockIdx.x * blockDim.x + threadIdx.x;
    if (idx >= SEQ * KVP) return;
    int s = idx >> 10;        // / 1024
    int c = idx & 1023;
    Vt[(size_t)c * SEQ + s] = V[idx];
}

// ---------------------------------------------------------------------------
// Causal flash attention: 1 wave per (32 query rows, head).
// Two 16-row query tiles per wave so each K/V fragment load feeds two WMMAs
// (halves K/V L2 traffic vs. 16 rows/wave). QK^T and P.V both run on
// v_wmma_f32_16x16x32_bf16; online softmax is kept in the WMMA C-layout
// (lane holds col n=lane%16, rows r+half*8), row statistics reduced across
// the 16-lane half-groups with shfl_xor(1,2,4,8). P is bounced through a
// 2 KB LDS tile to convert C-layout -> A-layout.
// ---------------------------------------------------------------------------
__global__ __launch_bounds__(32)
void flash_attn_kernel(const unsigned short* __restrict__ Q,
                       const unsigned short* __restrict__ Kc,
                       const unsigned short* __restrict__ Vt,
                       unsigned short* __restrict__ ctx) {
    __shared__ __attribute__((aligned(32))) unsigned short Plds[2][16][32];

    const int lane  = threadIdx.x & 31;
    const int hf    = lane >> 4;
    const int lm    = lane & 15;
    const int qbase = blockIdx.x * 32;
    const int h     = blockIdx.y;
    const int kv    = h >> 2;                 // GQA: 4 q-heads per kv-head

    // Q fragments: 2 row-tiles x 4 K-chunks (rows qbase+qt*16+lm, d 0..127).
    v16bf qF[2][4];
#pragma unroll
    for (int qt = 0; qt < 2; ++qt) {
        const unsigned short* qrow = Q + (size_t)(qbase + qt * 16 + lm) * HID + h * HD;
#pragma unroll
        for (int c = 0; c < 4; ++c)
            qF[qt][c] = make_afrag(qrow + c * 32 + hf * 16);
    }

    v8f O[2][8];
#pragma unroll
    for (int qt = 0; qt < 2; ++qt)
#pragma unroll
        for (int dt = 0; dt < 8; ++dt) O[qt][dt] = v8f{0.f,0.f,0.f,0.f,0.f,0.f,0.f,0.f};
    float mrow[2][8], lrow[2][8];
#pragma unroll
    for (int qt = 0; qt < 2; ++qt)
#pragma unroll
        for (int r = 0; r < 8; ++r) { mrow[qt][r] = -3.0e38f; lrow[qt][r] = 0.f; }

    const int nkb = (qbase + 31) / 32 + 1;    // key blocks up to the diagonal
    for (int kb = 0; kb < nkb; ++kb) {
        const int key0 = kb * 32;

        // ---- S = Q . K^T : 2 query tiles x 2 key tiles, K frags shared ----
        v8f s[2][2];
#pragma unroll
        for (int qt = 0; qt < 2; ++qt)
#pragma unroll
            for (int t = 0; t < 2; ++t) s[qt][t] = v8f{0.f,0.f,0.f,0.f,0.f,0.f,0.f,0.f};
#pragma unroll
        for (int c = 0; c < 4; ++c) {
            v16bf b0 = make_bfrag(Kc + (size_t)(key0 + lm) * KVP      + kv * HD + c * 32 + hf * 16);
            v16bf b1 = make_bfrag(Kc + (size_t)(key0 + 16 + lm) * KVP + kv * HD + c * 32 + hf * 16);
#pragma unroll
            for (int qt = 0; qt < 2; ++qt) {
                s[qt][0] = wmma_bf16(qF[qt][c], b0, s[qt][0]);
                s[qt][1] = wmma_bf16(qF[qt][c], b1, s[qt][1]);
            }
        }

        // ---- causal mask + online softmax update (per query tile) ----
        const int k0c = key0 + lm, k1c = key0 + 16 + lm;
        __syncthreads();     // previous iteration's P-fragment reads done
#pragma unroll
        for (int qt = 0; qt < 2; ++qt) {
#pragma unroll
            for (int r = 0; r < 8; ++r) {
                int qr = qbase + qt * 16 + hf * 8 + r;
                if (k0c > qr) s[qt][0][r] = -3.0e38f;
                if (k1c > qr) s[qt][1][r] = -3.0e38f;
            }
#pragma unroll
            for (int r = 0; r < 8; ++r) {
                float bm = fmaxf(s[qt][0][r], s[qt][1][r]);
                bm = fmaxf(bm, __shfl_xor(bm, 1, 32));
                bm = fmaxf(bm, __shfl_xor(bm, 2, 32));
                bm = fmaxf(bm, __shfl_xor(bm, 4, 32));
                bm = fmaxf(bm, __shfl_xor(bm, 8, 32));
                float nm    = fmaxf(mrow[qt][r], bm);
                float alpha = __expf(mrow[qt][r] - nm);
                float e0 = __expf(s[qt][0][r] - nm);
                float e1 = __expf(s[qt][1][r] - nm);
                float sum = e0 + e1;
                sum += __shfl_xor(sum, 1, 32);
                sum += __shfl_xor(sum, 2, 32);
                sum += __shfl_xor(sum, 4, 32);
                sum += __shfl_xor(sum, 8, 32);
                lrow[qt][r] = lrow[qt][r] * alpha + sum;
                mrow[qt][r] = nm;
                Plds[qt][hf * 8 + r][lm]      = f2bf(e0);
                Plds[qt][hf * 8 + r][16 + lm] = f2bf(e1);
#pragma unroll
                for (int dt = 0; dt < 8; ++dt) O[qt][dt][r] *= alpha;
            }
        }
        __syncthreads();     // P tiles visible in A-layout order

        // ---- C-layout P -> A-layout fragments ----
        v16bf pF0 = make_afrag(&Plds[0][lm][hf * 16]);
        v16bf pF1 = make_afrag(&Plds[1][lm][hf * 16]);

        // ---- O += P . V : V frags shared across both query tiles ----
#pragma unroll
        for (int dt = 0; dt < 8; ++dt) {
            v16bf bv = make_bfrag(Vt + (size_t)(kv * HD + dt * 16 + lm) * SEQ + key0 + hf * 16);
            O[0][dt] = wmma_bf16(pF0, bv, O[0][dt]);
            O[1][dt] = wmma_bf16(pF1, bv, O[1][dt]);
        }
    }

    // ---- epilogue: normalize and store bf16 context [s, h*128+d] ----
#pragma unroll
    for (int qt = 0; qt < 2; ++qt) {
        float invl[8];
#pragma unroll
        for (int r = 0; r < 8; ++r) invl[r] = 1.0f / lrow[qt][r];
#pragma unroll
        for (int dt = 0; dt < 8; ++dt)
#pragma unroll
            for (int r = 0; r < 8; ++r)
                ctx[(size_t)(qbase + qt * 16 + hf * 8 + r) * HID + h * HD + dt * 16 + lm] =
                    f2bf(O[qt][dt][r] * invl[r]);
    }
}

// ---------------------------------------------------------------------------
// Host-side orchestration
// ---------------------------------------------------------------------------
extern "C" void kernel_launch(void* const* d_in, const int* in_sizes, int n_in,
                              void* d_out, int out_size, void* d_ws, size_t ws_size,
                              hipStream_t stream) {
    (void)in_sizes; (void)n_in; (void)out_size; (void)ws_size;

    const float* hs = (const float*)d_in[0];
    const float* Wq = (const float*)d_in[1];
    const float* Wk = (const float*)d_in[2];
    const float* Wv = (const float*)d_in[3];
    const float* Wo = (const float*)d_in[4];
    float* out = (float*)d_out;

    char* ws = (char*)d_ws;
    size_t off = 0;
    auto carve = [&](size_t bytes) -> unsigned short* {
        unsigned short* p = (unsigned short*)(ws + off);
        off += (bytes + 255) & ~(size_t)255;
        return p;
    };
    unsigned short* Xb  = carve((size_t)SEQ * HID * 2);
    unsigned short* Wqb = carve((size_t)HID * HID * 2);
    unsigned short* Wkb = carve((size_t)KVP * HID * 2);
    unsigned short* Wvb = carve((size_t)KVP * HID * 2);
    unsigned short* Wob = carve((size_t)HID * HID * 2);
    unsigned short* Qb  = carve((size_t)SEQ * HID * 2);
    unsigned short* Kb  = carve((size_t)SEQ * KVP * 2);
    unsigned short* Vb  = carve((size_t)SEQ * KVP * 2);
    unsigned short* Vtb = carve((size_t)SEQ * KVP * 2);
    unsigned short* Ctx = carve((size_t)SEQ * HID * 2);

    auto cvt = [&](const float* s, unsigned short* d, int n) {
        int n8 = n / 8;
        cvt_bf16_kernel<<<(n8 + 255) / 256, 256, 0, stream>>>(s, d, n8);
    };
    cvt(hs, Xb,  SEQ * HID);
    cvt(Wq, Wqb, HID * HID);
    cvt(Wk, Wkb, KVP * HID);
    cvt(Wv, Wvb, KVP * HID);
    cvt(Wo, Wob, HID * HID);

    // Projections: Q/K/V = X . W^T
    gemm_bf16_wmma<unsigned short>
        <<<dim3(SEQ / 128, HID / 128), 128, 0, stream>>>(Xb, Wqb, Qb, SEQ, HID, HID);
    gemm_bf16_wmma<unsigned short>
        <<<dim3(SEQ / 128, KVP / 128), 128, 0, stream>>>(Xb, Wkb, Kb, SEQ, KVP, HID);
    gemm_bf16_wmma<unsigned short>
        <<<dim3(SEQ / 128, KVP / 128), 128, 0, stream>>>(Xb, Wvb, Vb, SEQ, KVP, HID);

    // RoPE (fold 1/sqrt(HD) into Q)
    const float scale = 0.08838834764831845f;
    {
        int totq = SEQ * NH * 64;
        rope_kernel<<<(totq + 255) / 256, 256, 0, stream>>>(Qb, NH, HID, scale, totq);
        int totk = SEQ * NKV * 64;
        rope_kernel<<<(totk + 255) / 256, 256, 0, stream>>>(Kb, NKV, KVP, 1.0f, totk);
    }

    // Transpose V for key-contiguous B-fragments in P.V
    transpose_v_kernel<<<(SEQ * KVP + 255) / 256, 256, 0, stream>>>(Vb, Vtb);

    // Causal flash attention (32 query rows per wave)
    flash_attn_kernel<<<dim3(SEQ / 32, NH), 32, 0, stream>>>(Qb, Kb, Vtb, Ctx);

    // Output projection -> fp32 result
    gemm_bf16_wmma<float>
        <<<dim3(SEQ / 128, HID / 128), 128, 0, stream>>>(Ctx, Wob, out, SEQ, HID, HID);
}
